// MaskCrossAttn_7799660609558
// MI455X (gfx1250) — compile-verified, matrix operand order checked by reading, not run on hardware
//
#include <hip/hip_runtime.h>
#include <hip/hip_bf16.h>
#include <stdint.h>

typedef _Float16 h16;
typedef __attribute__((ext_vector_type(16))) _Float16 v16h;
typedef __attribute__((ext_vector_type(8)))  _Float16 v8h;
typedef __attribute__((ext_vector_type(8)))  float    v8f;
typedef __attribute__((ext_vector_type(4)))  int      v4i;

#define BB    8
#define CC    192
#define HWN   16384          // 128*128
#define MROWS (BB*HWN)       // 131072
#define NHEAD 8
#define DH    24             // CC/NHEAD
#define HIDN  384            // 2*CC

#ifndef __has_builtin
#define __has_builtin(x) 0
#endif
#if __has_builtin(__builtin_amdgcn_global_load_async_to_lds_b128)
#define HAVE_ASYNC_LDS 1
#else
#define HAVE_ASYNC_LDS 0
#endif

// Address-space-qualified pointee typedefs (attribute must bind to pointee).
typedef __attribute__((address_space(1))) v4i g_v4i;   // global int4
typedef __attribute__((address_space(3))) v4i l_v4i;   // LDS int4

// 16B global -> LDS copy: async LDS-DMA (ASYNCcnt) when available.
__device__ __forceinline__ void copy16_g2l(const h16* g, h16* l) {
#if HAVE_ASYNC_LDS
    __builtin_amdgcn_global_load_async_to_lds_b128((g_v4i*)g, (l_v4i*)l, 0, 0);
#else
    *(v8h*)l = *(const v8h*)g;
#endif
}

__device__ __forceinline__ void async_wait_all() {
#if HAVE_ASYNC_LDS
#if __has_builtin(__builtin_amdgcn_s_wait_asynccnt)
    __builtin_amdgcn_s_wait_asynccnt(0);
#else
    asm volatile("s_wait_asynccnt 0" ::: "memory");
#endif
#endif
}

// Intra-wave LDS store->load ordering (per-wave private tiles; no barrier).
__device__ __forceinline__ void lds_fence_wave() {
#if __has_builtin(__builtin_amdgcn_s_wait_dscnt)
    __builtin_amdgcn_s_wait_dscnt(0);
#else
    asm volatile("s_wait_dscnt 0" ::: "memory");
#endif
}

// ---------------------------------------------------------------- utilities

__global__ void wcvt_kernel(const float* __restrict__ src, h16* __restrict__ dst, int n) {
    int i = blockIdx.x * 256 + threadIdx.x;
    if (i < n) dst[i] = (h16)src[i];
}

// LayerNorm over channels + mask prep. One wave per pixel-row; transposed
// (channel-strided) reads from [B,C,H,W], writes [M,C] f16 for GEMM.
__global__ __launch_bounds__(256)
void ln1_kernel(const float* __restrict__ img, const float* __restrict__ mask,
                const float* __restrict__ w, const float* __restrict__ bvec,
                h16* __restrict__ xn, h16* __restrict__ xq) {
    int m    = blockIdx.x * 8 + (threadIdx.x >> 5);
    int lane = threadIdx.x & 31;
    int b = m >> 14, np = m & (HWN - 1);
    size_t base = (size_t)b * CC * HWN + np;
    float v[6], s = 0.f, s2 = 0.f;
#pragma unroll
    for (int j = 0; j < 6; ++j) {
        int c = lane + 32 * j;
        float x = img[base + (size_t)c * HWN];
        v[j] = x; s += x; s2 += x * x;
    }
#pragma unroll
    for (int o = 16; o > 0; o >>= 1) { s += __shfl_xor(s, o, 32); s2 += __shfl_xor(s2, o, 32); }
    float mu = s * (1.f / CC);
    float var = s2 * (1.f / CC) - mu * mu;
    float rs = rsqrtf(var + 1e-5f);
#pragma unroll
    for (int j = 0; j < 6; ++j) {
        int c = lane + 32 * j;
        float xv = (v[j] - mu) * rs * w[c] + bvec[c];
        float mk = fmaxf(mask[base + (size_t)c * HWN], 0.1f);
        xn[(size_t)m * CC + c] = (h16)xv;
        xq[(size_t)m * CC + c] = (h16)(xv * mk);
    }
}

// LayerNorm over channels of fused [M,C] f32 -> xln f16
__global__ __launch_bounds__(256)
void ln2_kernel(const float* __restrict__ fused, const float* __restrict__ w,
                const float* __restrict__ bvec, h16* __restrict__ xln) {
    int m    = blockIdx.x * 8 + (threadIdx.x >> 5);
    int lane = threadIdx.x & 31;
    const float* rowp = fused + (size_t)m * CC;
    float v[6], s = 0.f, s2 = 0.f;
#pragma unroll
    for (int j = 0; j < 6; ++j) {
        float x = rowp[lane + 32 * j];
        v[j] = x; s += x; s2 += x * x;
    }
#pragma unroll
    for (int o = 16; o > 0; o >>= 1) { s += __shfl_xor(s, o, 32); s2 += __shfl_xor(s2, o, 32); }
    float mu = s * (1.f / CC);
    float var = s2 * (1.f / CC) - mu * mu;
    float rs = rsqrtf(var + 1e-5f);
#pragma unroll
    for (int j = 0; j < 6; ++j) {
        int c = lane + 32 * j;
        xln[(size_t)m * CC + c] = (h16)((v[j] - mu) * rs * w[c] + bvec[c]);
    }
}

// Build a 16x32 A/B fragment (v16h) from an LDS row: halves at cols
// [hi..hi+7] and [16+hi..23+hi], hi = (lane<16)?0:8 per CDNA5 WMMA layout.
__device__ __forceinline__ v16h frag_from_lds(const h16* rowbase, int hi) {
    v8h lo = *(const v8h*)(rowbase + hi);
    v8h up = *(const v8h*)(rowbase + 16 + hi);
    v16h r;
#pragma unroll
    for (int i = 0; i < 8; ++i) { r[i] = lo[i]; r[8 + i] = up[i]; }
    return r;
}

// ------------------------------------------------------- generic WMMA GEMM
// D[M,N] = A[M,K] * W[N,K]^T (+ epilogue).  Block tile 128x64, 8 waves,
// each wave 32x32 (= 2x2 v_wmma_f32_16x16x32_f16 tiles), K stepped by 32.
// MODE 0: store f16                      (q/k/v projections)
// MODE 1: + image residual, store f32    (fused = image + Wo@attn)
// MODE 2: + bias, exact GELU, store f16  (fc1)
// MODE 3: + bias + fused residual, transpose-store f32 to [B,C,H,W] (fc2)
template<int MODE>
__global__ __launch_bounds__(256)
void gemm_kernel(const h16* __restrict__ A, const h16* __restrict__ Wt, int K,
                 void* __restrict__ outp, const float* __restrict__ bias,
                 const float* __restrict__ extra) {
    __shared__ h16 As[128 * 40];
    __shared__ h16 Bs[64 * 40];
    const int tid  = threadIdx.x;
    const int lane = tid & 31;
    const int wv   = tid >> 5;
    const int wm   = wv & 3, wn = wv >> 2;
    const int m0   = blockIdx.x * 128;
    const int n0   = blockIdx.y * 64;
    const int N    = (int)gridDim.y * 64;
    const int hi   = (lane < 16) ? 0 : 8;

    v8f acc[2][2] = {};

    for (int k0 = 0; k0 < K; k0 += 32) {
        // stage A tile (128x32 halves = 512 b128 chunks) and B tile (64x32)
        // through the async LDS-DMA path when available
#pragma unroll
        for (int r = 0; r < 2; ++r) {
            int ch = tid + r * 256;
            int row = ch >> 2, c8 = (ch & 3) * 8;
            copy16_g2l(A + (size_t)(m0 + row) * K + k0 + c8, As + row * 40 + c8);
        }
        {
            int row = tid >> 2, c8 = (tid & 3) * 8;
            copy16_g2l(Wt + (size_t)(n0 + row) * K + k0 + c8, Bs + row * 40 + c8);
        }
        if (k0 + 32 < K)
            __builtin_prefetch(A + (size_t)(m0 + (tid >> 1)) * K + k0 + 32, 0, 1);
        async_wait_all();
        __syncthreads();

        v16h af[2], bf[2];
#pragma unroll
        for (int t = 0; t < 2; ++t)
            af[t] = frag_from_lds(As + (wm * 32 + t * 16 + (lane & 15)) * 40, hi);
#pragma unroll
        for (int t = 0; t < 2; ++t)
            bf[t] = frag_from_lds(Bs + (wn * 32 + t * 16 + (lane & 15)) * 40, hi);
#pragma unroll
        for (int mt = 0; mt < 2; ++mt)
#pragma unroll
            for (int nt = 0; nt < 2; ++nt)
                acc[mt][nt] = __builtin_amdgcn_wmma_f32_16x16x32_f16(
                    false, af[mt], false, bf[nt], (short)0, acc[mt][nt], false, false);
        __syncthreads();
    }

    const int mo = (lane < 16) ? 0 : 8;
#pragma unroll
    for (int mt = 0; mt < 2; ++mt)
#pragma unroll
        for (int nt = 0; nt < 2; ++nt) {
#pragma unroll
            for (int j = 0; j < 8; ++j) {
                int gm = m0 + wm * 32 + mt * 16 + mo + j;
                int gn = n0 + wn * 32 + nt * 16 + (lane & 15);
                float val = acc[mt][nt][j];
                if (MODE == 0) {
                    ((h16*)outp)[(size_t)gm * N + gn] = (h16)val;
                } else if (MODE == 1) {
                    int b = gm >> 14, np = gm & (HWN - 1);
                    val += extra[((size_t)b * CC + gn) * HWN + np];
                    ((float*)outp)[(size_t)gm * N + gn] = val;
                } else if (MODE == 2) {
                    val += bias[gn];
                    float g = 0.5f * val * (1.0f + erff(val * 0.70710678118f));
                    ((h16*)outp)[(size_t)gm * N + gn] = (h16)g;
                } else {
                    int b = gm >> 14, np = gm & (HWN - 1);
                    val += bias[gn] + extra[(size_t)gm * CC + gn];
                    ((float*)outp)[((size_t)b * CC + gn) * HWN + np] = val;
                }
            }
        }
}

// -------------------------------------------------- channel attention core
// One block per (b,head). Q/K/V stored as [M,C] f16 (row = pixel).
// attn = softmax( (Qs^T n) (Ks n)^T * temp ), out = attn @ Vs^T ; dh=24 pad 32.
__global__ __launch_bounds__(256)
void attn_kernel(const h16* __restrict__ Qb, const h16* __restrict__ Kb,
                 const h16* __restrict__ Vb, const float* __restrict__ temp,
                 h16* __restrict__ AO) {
    __shared__ h16   qT[8][32 * 40];
    __shared__ h16   kT[8][32 * 40];
    __shared__ float attnS[32 * 32];
    __shared__ h16   attnh[32 * 32];
    __shared__ float red[256];
    __shared__ float rq[32], rk[32];

    const int tid = threadIdx.x, lane = tid & 31, wv = tid >> 5;
    const int bh = blockIdx.x;
    const int b = bh >> 3, head = bh & 7;
    const size_t bHW = (size_t)b * HWN;
    const int off = head * DH;
    const int hi  = (lane < 16) ? 0 : 8;

    for (int i = tid; i < 8 * 32 * 40; i += 256) { (&qT[0][0])[i] = (h16)0.f; (&kT[0][0])[i] = (h16)0.f; }
    for (int i = tid; i < 32 * 32; i += 256)     { attnS[i] = 0.f; attnh[i] = (h16)0.f; }
    __syncthreads();

    // L2 norms over hw for each channel column of this head's Q and K slices
    for (int c = 0; c < DH; ++c) {
        float sq = 0.f, sk = 0.f;
        for (int n = tid; n < HWN; n += 256) {
            size_t idx = (bHW + n) * CC + off + c;
            float qv = (float)Qb[idx], kv = (float)Kb[idx];
            sq += qv * qv; sk += kv * kv;
        }
        red[tid] = sq; __syncthreads();
        for (int s = 128; s > 0; s >>= 1) { if (tid < s) red[tid] += red[tid + s]; __syncthreads(); }
        if (tid == 0) rq[c] = 1.f / fmaxf(sqrtf(red[0]), 1e-12f);
        __syncthreads();
        red[tid] = sk; __syncthreads();
        for (int s = 128; s > 0; s >>= 1) { if (tid < s) red[tid] += red[tid + s]; __syncthreads(); }
        if (tid == 0) rk[c] = 1.f / fmaxf(sqrtf(red[0]), 1e-12f);
        __syncthreads();
    }
    if (tid < 8) { rq[24 + tid] = 0.f; rk[24 + tid] = 0.f; }
    __syncthreads();

    // Phase A: attn[c,d] += sum_n qn[c,n] * kn[d,n]  via WMMA over n-chunks of 32.
    // Per-wave private LDS transpose tiles -> only intra-wave dscnt ordering.
    v8f acc[2][2] = {};
    for (int it = 0; it < 64; ++it) {
        int n0 = (it * 8 + wv) * 32;
        int n = n0 + lane;
        const h16* qrow = Qb + (bHW + n) * CC + off;
        const h16* krow = Kb + (bHW + n) * CC + off;
        v8h q0 = *(const v8h*)qrow, q1 = *(const v8h*)(qrow + 8), q2 = *(const v8h*)(qrow + 16);
        v8h k0 = *(const v8h*)krow, k1 = *(const v8h*)(krow + 8), k2 = *(const v8h*)(krow + 16);
        h16* qt = qT[wv]; h16* kt = kT[wv];
#pragma unroll
        for (int i = 0; i < 8; ++i) {
            qt[i * 40 + lane] = q0[i]; qt[(8 + i) * 40 + lane] = q1[i]; qt[(16 + i) * 40 + lane] = q2[i];
            kt[i * 40 + lane] = k0[i]; kt[(8 + i) * 40 + lane] = k1[i]; kt[(16 + i) * 40 + lane] = k2[i];
        }
        lds_fence_wave();
        v16h af[2], bf[2];
#pragma unroll
        for (int t = 0; t < 2; ++t) {
            int c = t * 16 + (lane & 15);
            af[t] = frag_from_lds(qt + c * 40, hi);
            bf[t] = frag_from_lds(kt + c * 40, hi);
            h16 r0 = (h16)rq[c], r1 = (h16)rk[c];
            af[t] = af[t] * r0;                 // v_pk_mul_f16, no f32 round-trip
            bf[t] = bf[t] * r1;
        }
#pragma unroll
        for (int mt = 0; mt < 2; ++mt)
#pragma unroll
            for (int nt = 0; nt < 2; ++nt)
                acc[mt][nt] = __builtin_amdgcn_wmma_f32_16x16x32_f16(
                    false, af[mt], false, bf[nt], (short)0, acc[mt][nt], false, false);
        lds_fence_wave();
    }
#pragma unroll
    for (int mt = 0; mt < 2; ++mt)
#pragma unroll
        for (int nt = 0; nt < 2; ++nt)
#pragma unroll
            for (int j = 0; j < 8; ++j) {
                int m = mt * 16 + ((lane < 16) ? 0 : 8) + j;
                int n = nt * 16 + (lane & 15);
                atomicAdd(&attnS[m * 32 + n], acc[mt][nt][j]);
            }
    __syncthreads();

    // softmax over d (24 valid) with temperature
    if (tid < 24) {
        float ts = temp[head];
        float row[24]; float mx = -1e30f;
#pragma unroll
        for (int d = 0; d < 24; ++d) { row[d] = attnS[tid * 32 + d] * ts; mx = fmaxf(mx, row[d]); }
        float s = 0.f;
#pragma unroll
        for (int d = 0; d < 24; ++d) { row[d] = __expf(row[d] - mx); s += row[d]; }
        float inv = 1.f / s;
#pragma unroll
        for (int d = 0; d < 24; ++d) attnh[tid * 32 + d] = (h16)(row[d] * inv);
    }
    __syncthreads();

    // Phase B: out[c,n] = sum_d attn[c,d] * v[d,n], tiles of 16 pixels
    v16h af2[2];
#pragma unroll
    for (int t = 0; t < 2; ++t)
        af2[t] = frag_from_lds(attnh + (t * 16 + (lane & 15)) * 32, hi);
    for (int t8 = wv; t8 < HWN / 16; t8 += 8) {
        int nb = t8 * 16;
        int n = nb + (lane & 15);
        const h16* vrow = Vb + (bHW + n) * CC + off;
        v8h b0 = *(const v8h*)(vrow + hi);
        v8h b1 = {};
        if (lane < 16) b1 = *(const v8h*)(vrow + 16);   // d=16..23 real, 24..31 pad=0
        v16h bf;
#pragma unroll
        for (int i = 0; i < 8; ++i) { bf[i] = b0[i]; bf[8 + i] = b1[i]; }
#pragma unroll
        for (int mt = 0; mt < 2; ++mt) {
            v8f z = {};
            v8f d = __builtin_amdgcn_wmma_f32_16x16x32_f16(
                false, af2[mt], false, bf, (short)0, z, false, false);
            int cbase = mt * 16 + ((lane < 16) ? 0 : 8);
            if (cbase < 24) {
                v8h ov;
#pragma unroll
                for (int j = 0; j < 8; ++j) ov[j] = (h16)d[j];
                *(v8h*)(AO + (bHW + n) * CC + off + cbase) = ov;
            }
        }
    }
}

// ---------------------------------------------------------------- launcher

extern "C" void kernel_launch(void* const* d_in, const int* in_sizes, int n_in,
                              void* d_out, int out_size, void* d_ws, size_t ws_size,
                              hipStream_t stream) {
    const float* image = (const float*)d_in[0];
    const float* mask  = (const float*)d_in[1];
    const float* n1_w  = (const float*)d_in[2];
    const float* n1_b  = (const float*)d_in[3];
    const float* Wq    = (const float*)d_in[4];
    const float* Wk    = (const float*)d_in[5];
    const float* Wv    = (const float*)d_in[6];
    const float* Wo    = (const float*)d_in[7];
    const float* temp  = (const float*)d_in[8];
    const float* n2_w  = (const float*)d_in[9];
    const float* n2_b  = (const float*)d_in[10];
    const float* fc1_w = (const float*)d_in[11];
    const float* fc1_b = (const float*)d_in[12];
    const float* fc2_w = (const float*)d_in[13];
    const float* fc2_b = (const float*)d_in[14];
    float* out = (float*)d_out;

    char* p = (char*)d_ws;
    h16* wqh = (h16*)p; p += (size_t)CC * CC * 2;
    h16* wkh = (h16*)p; p += (size_t)CC * CC * 2;
    h16* wvh = (h16*)p; p += (size_t)CC * CC * 2;
    h16* woh = (h16*)p; p += (size_t)CC * CC * 2;
    h16* f1h = (h16*)p; p += (size_t)HIDN * CC * 2;
    h16* f2h = (h16*)p; p += (size_t)CC * HIDN * 2;
    p = (char*)(((uintptr_t)p + 255) & ~(uintptr_t)255);
    const size_t MC = (size_t)MROWS * CC;
    h16* xn = (h16*)p; p += MC * 2;           // later reused as xln
    h16* xq = (h16*)p; p += MC * 2;           // later reused as attn-out
    h16* qb = (h16*)p; p += MC * 2;
    h16* kb = (h16*)p; p += MC * 2;           // kb..vb contiguous -> hmid [M,384]
    h16* vb = (h16*)p; p += MC * 2;
    float* fused = (float*)p; p += MC * 4;
    (void)ws_size; (void)in_sizes; (void)n_in; (void)out_size; (void)vb;

    const int nW = CC * CC, nF = HIDN * CC;
    wcvt_kernel<<<(nW + 255) / 256, 256, 0, stream>>>(Wq, wqh, nW);
    wcvt_kernel<<<(nW + 255) / 256, 256, 0, stream>>>(Wk, wkh, nW);
    wcvt_kernel<<<(nW + 255) / 256, 256, 0, stream>>>(Wv, wvh, nW);
    wcvt_kernel<<<(nW + 255) / 256, 256, 0, stream>>>(Wo, woh, nW);
    wcvt_kernel<<<(nF + 255) / 256, 256, 0, stream>>>(fc1_w, f1h, nF);
    wcvt_kernel<<<(nF + 255) / 256, 256, 0, stream>>>(fc2_w, f2h, nF);

    ln1_kernel<<<MROWS / 8, 256, 0, stream>>>(image, mask, n1_w, n1_b, xn, xq);

    dim3 gC(MROWS / 128, CC / 64);
    dim3 gH(MROWS / 128, HIDN / 64);
    gemm_kernel<0><<<gC, 256, 0, stream>>>(xq, wqh, CC, qb, nullptr, nullptr);
    gemm_kernel<0><<<gC, 256, 0, stream>>>(xn, wkh, CC, kb, nullptr, nullptr);
    gemm_kernel<0><<<gC, 256, 0, stream>>>(xn, wvh, CC, vb, nullptr, nullptr);

    attn_kernel<<<BB * NHEAD, 256, 0, stream>>>(qb, kb, vb, temp, xq);

    gemm_kernel<1><<<gC, 256, 0, stream>>>(xq, woh, CC, fused, nullptr, image);
    ln2_kernel<<<MROWS / 8, 256, 0, stream>>>(fused, n2_w, n2_b, xn);
    gemm_kernel<2><<<gH, 256, 0, stream>>>(xn, f1h, CC, kb, fc1_b, nullptr);   // hmid -> kb..vb
    gemm_kernel<3><<<gC, 256, 0, stream>>>(kb, f2h, HIDN, out, fc2_b, fused);
}
